// Encoder_84164179133136
// MI455X (gfx1250) — matrix operand order repaired and checked by reading, not compiled, not run on hardware
//
#include <hip/hip_runtime.h>
#include <hip/hip_bf16.h>

// ---------------- problem constants ----------------
#define Bc    16
#define Sc    256
#define Dc    41
#define Lc    (Sc*Dc)     // 10496, multiple of 128
#define LATc  128
#define NHc   4
#define DHc   32
#define KSPLIT 8          // MAB0 flash split-K factor (Lc/KSPLIT = 1312, mult of 32)

// ---------------- WMMA types ----------------
typedef __attribute__((ext_vector_type(16))) __bf16        v16bf;
typedef __attribute__((ext_vector_type(8)))  float         v8f;
typedef __attribute__((ext_vector_type(8)))  unsigned short ush8;

union V16u { v16bf v; unsigned short s[16]; ush8 h[2]; };
union V8u  { ush8 v; unsigned short s[8]; };

// reuse hints must be literal constants at the builtin call site
template<bool RA, bool RB>
__device__ __forceinline__ v8f wmma_bf16(v16bf a, v16bf b, v8f c){
  return __builtin_amdgcn_wmma_f32_16x16x32_bf16(false, a, false, b, (short)0, c, RA, RB);
}

__device__ __forceinline__ unsigned short f2bf(float f){
  union { float f; unsigned u; } x; x.f = f;
  unsigned u = x.u;
  u += 0x7fffu + ((u >> 16) & 1u);       // round-to-nearest-even
  return (unsigned short)(u >> 16);
}
__device__ __forceinline__ float bf2f(unsigned short h){
  union { unsigned u; float f; } x; x.u = ((unsigned)h) << 16; return x.f;
}

// ---------------- utility kernels ----------------
__global__ void zero_f32_kernel(float* __restrict__ p, long n){
  long i = (long)blockIdx.x * blockDim.x + threadIdx.x;
  if (i < n) p[i] = 0.0f;
}

__global__ void f32_to_bf16_kernel(const float* __restrict__ s,
                                   unsigned short* __restrict__ d, long n){
  long i = (long)blockIdx.x * blockDim.x + threadIdx.x;
  if (i < n) d[i] = f2bf(s[i]);
}

// Convert 128x128 f32 weight matrices to bf16 in WMMA B-fragment order:
// element (kc,n,lane,i) = W[kc*32 + (lane>>4)*16 + i][n*16 + (lane&15)]
__global__ void swizzle_w_kernel(const float* __restrict__ src,
                                 unsigned short* __restrict__ dst, long total){
  long t = (long)blockIdx.x * blockDim.x + threadIdx.x;
  if (t >= total) return;
  int mat = (int)(t >> 14);
  int o   = (int)(t & 16383);
  int i    = o & 15;
  int lane = (o >> 4) & 31;
  int n    = (o >> 9) & 7;
  int kc   = o >> 12;
  int k    = kc*32 + ((lane >> 4) << 4) + i;
  int col  = (n << 4) + (lane & 15);
  dst[t] = f2bf(src[((size_t)mat << 14) + (size_t)k*128 + col]);
}

// ---------------- stable compaction (argsort(1-mask), stable) ----------------
__global__ __launch_bounds__(256) void compact_kernel(
    const int* __restrict__ mask, int* __restrict__ idx, float* __restrict__ mk)
{
  __shared__ int ssum[256];
  __shared__ int scarry;
  int b = blockIdx.x, t = threadIdx.x;
  if (t == 0) scarry = 0;
  __syncthreads();
  for (int base = 0; base < Lc; base += 256){
    int l = base + t;
    int m = (l < Lc) ? mask[b*Lc + l] : 0;
    ssum[t] = m;
    __syncthreads();
    #pragma unroll
    for (int off = 1; off < 256; off <<= 1){
      int v = (t >= off) ? ssum[t - off] : 0;
      __syncthreads();
      ssum[t] += v;
      __syncthreads();
    }
    int incl  = ssum[t];
    int carry = scarry;
    __syncthreads();
    if (m){
      int dest = carry + incl - 1;           // stable partition
      idx[b*Lc + dest] = l;
      mk [b*Lc + dest] = 1.0f;
    }
    if (t == 255) scarry = carry + ssum[255];
    __syncthreads();
  }
}

// ---------------- fused gather + input projection ----------------
__global__ __launch_bounds__(128) void embed_kernel(
    const float* __restrict__ time_x, const float* __restrict__ value_x,
    const int* __restrict__ idx, const float* __restrict__ mk,
    const float* __restrict__ Wi, const float* __restrict__ bi,
    float* __restrict__ Zf, unsigned short* __restrict__ Zb)
{
  int row = blockIdx.x;                 // b*Lc + dest
  int j   = threadIdx.x;                // latent channel
  int b   = row / Lc;
  float z = 0.0f;
  if (mk[row] != 0.0f){
    int l = idx[row];
    int s = l / Dc;
    int c = l - s * Dc;
    float t = time_x[b*Sc + s];
    float u = value_x[(size_t)b*Lc + l];
    z = Wi[(size_t)c*LATc + j] + t*Wi[(size_t)Dc*LATc + j]
      + u*Wi[(size_t)(Dc+1)*LATc + j] + bi[j];
    z = fmaxf(z, 0.0f);
  }
  size_t o = (size_t)row * LATc + j;
  Zf[o] = z;
  Zb[o] = f2bf(z);
}

// ---------------- 128-wide bf16 WMMA GEMM, Y = A(R,128) @ W(128,128) ----------------
// Wf is in fragment order: each B-frag = 2 x global b128 loads.
// a_mask: AND-mask on the A row index (127 = broadcast induced points, ~0 = identity)
// mode 0: outB = bf16(acc + bias)          (vt_n>0: store transposed [b][h][d][key])
// mode 1: outB = bf16(Res + relu(acc + bias))
// mode 2: z = Zf + mk[row]*(Res + relu(acc+bias)); Zf = z; outB = bf16(z)
__global__ __launch_bounds__(256) void gemm128_kernel(
    const unsigned short* __restrict__ A, int a_mask,
    const unsigned short* __restrict__ Wf, const float* __restrict__ bias,
    int mode,
    unsigned short* __restrict__ outB,
    const unsigned short* __restrict__ Res,
    const float* __restrict__ mk,
    float* __restrict__ Zf,
    int R, int vt_n)
{
  int tid  = threadIdx.x;
  int wave = tid >> 5, lane = tid & 31;
  int row0 = blockIdx.x * 128 + wave * 16;
  int aM   = lane & 15, aK0 = (lane >> 4) * 16;
  int bN   = lane & 15;

  const v8f vzero = {0.f,0.f,0.f,0.f,0.f,0.f,0.f,0.f};
  v8f acc[8];
  #pragma unroll
  for (int n = 0; n < 8; ++n) acc[n] = vzero;

  int ar = (row0 + aM) & a_mask;         // no integer division in the hot loop
  const unsigned short* arow = A + (size_t)ar * 128;

  #pragma unroll
  for (int kc = 0; kc < 4; ++kc){        // K chunks of 32
    const unsigned short* ap = arow + kc * 32;
    V16u av;
    av.h[0] = *(const ush8*)(ap + aK0);
    av.h[1] = *(const ush8*)(ap + aK0 + 8);
    // A fragment shared by 8 consecutive WMMAs -> reuse-A hint on the first 7
    #pragma unroll
    for (int n = 0; n < 7; ++n){
      const unsigned short* bp = Wf + (((size_t)(kc*8 + n)*32 + lane) << 4);
      V16u bv_;
      bv_.h[0] = *(const ush8*)(bp);
      bv_.h[1] = *(const ush8*)(bp + 8);
      acc[n] = wmma_bf16<true, false>(av.v, bv_.v, acc[n]);
    }
    {
      const unsigned short* bp = Wf + (((size_t)(kc*8 + 7)*32 + lane) << 4);
      V16u bv_;
      bv_.h[0] = *(const ush8*)(bp);
      bv_.h[1] = *(const ush8*)(bp + 8);
      acc[7] = wmma_bf16<false, false>(av.v, bv_.v, acc[7]);
    }
  }

  if (vt_n > 0){
    // transposed store: rows are attention keys; 8 consecutive keys per lane
    int keyrow0 = row0 + 8*(lane >> 4);
    int bb  = keyrow0 / vt_n;
    int key = keyrow0 - bb * vt_n;
    #pragma unroll
    for (int n = 0; n < 8; ++n){
      int col = n*16 + bN;
      float bs = bias[col];
      V8u pk;
      #pragma unroll
      for (int r = 0; r < 8; ++r) pk.s[r] = f2bf(acc[n][r] + bs);
      int hh = col >> 5, dd = col & 31;
      *(ush8*)(outB + ((size_t)(bb*NHc + hh)*DHc + dd) * (size_t)vt_n + key) = pk.v;
    }
  } else {
    #pragma unroll
    for (int n = 0; n < 8; ++n){
      #pragma unroll
      for (int r = 0; r < 8; ++r){
        int row = row0 + r + 8*(lane >> 4);
        int col = n*16 + bN;
        size_t o = (size_t)row * 128 + col;
        float v = acc[n][r] + bias[col];
        if (mode == 0){
          outB[o] = f2bf(v);
        } else if (mode == 1){
          float ov = bf2f(Res[o]) + fmaxf(v, 0.0f);
          outB[o] = f2bf(ov);
        } else {
          float ov = bf2f(Res[o]) + fmaxf(v, 0.0f);
          float z  = Zf[o] + mk[row] * ov;
          Zf[o]   = z;
          outB[o] = f2bf(z);
        }
      }
    }
  }
}

// ---------------- flash-attention MAB core (wave32 WMMA) ----------------
// per wave: 16 queries; online softmax over 32-key blocks.
// Vt is transposed [b][h][d][key] (stride nK).
// Single-pass (pacc==null): O = Q + softmax(Q.K^T/sqrt(128), mask)·V
// Split-K   (pacc!=null): write unnormalized partial acc + (m,l) per split.
__global__ __launch_bounds__(256) void attn_kernel(
    const unsigned short* __restrict__ Q,
    const unsigned short* __restrict__ Km,
    const unsigned short* __restrict__ Vt,
    const float* __restrict__ mkv,        // null -> no key mask
    unsigned short* __restrict__ O,       // single-pass output
    float* __restrict__ pacc,             // split partials (or null)
    float* __restrict__ pml,              // split (m,l) pairs (or null)
    int nQ, int nK, int nQtiles, int kChunk)
{
  __shared__ unsigned short sP[8][16][32];   // per-wave P (C-layout -> A-layout)
  int tid  = threadIdx.x;
  int wave = tid >> 5, lane = tid & 31;
  int blk  = blockIdx.x;
  int h    = blk & 3;
  int rest = blk >> 2;
  int qt   = rest % nQtiles;
  int rest2= rest / nQtiles;
  int b    = rest2 % Bc;
  int split= rest2 / Bc;

  int k0   = split * kChunk;
  int kend = k0 + kChunk; if (kend > nK) kend = nK;

  int aK0 = (lane >> 4) * 16;
  int bN  = lane & 15, bK0 = (lane >> 4) * 16;

  // A-fragment of Q (dh = 32 = one WMMA K)
  int qrowA = qt*128 + wave*16 + (lane & 15);
  size_t qa_off = ((size_t)(b*nQ + qrowA)) * LATc + h*DHc;
  V16u qa;
  qa.h[0] = *(const ush8*)(Q + qa_off + aK0);
  qa.h[1] = *(const ush8*)(Q + qa_off + aK0 + 8);

  const unsigned short* vrow0 = Vt + ((size_t)(b*NHc + h)*DHc + bN)      * (size_t)nK;
  const unsigned short* vrow1 = Vt + ((size_t)(b*NHc + h)*DHc + 16 + bN) * (size_t)nK;

  const v8f vzero = {0.f,0.f,0.f,0.f,0.f,0.f,0.f,0.f};
  v8f acc0 = vzero, acc1 = vzero;
  float mrow[8], lrow[8];
  #pragma unroll
  for (int r = 0; r < 8; ++r){ mrow[r] = -3.0e38f; lrow[r] = 0.0f; }
  const float scale = 0.0883883476483184f;   // 1/sqrt(128)

  for (int kb = k0; kb < kend; kb += 32){
    V16u kf0, kf1;
    size_t kro = ((size_t)(b*nK + kb +      bN)) * LATc + h*DHc + bK0;
    size_t kr1 = ((size_t)(b*nK + kb + 16 + bN)) * LATc + h*DHc + bK0;
    kf0.h[0] = *(const ush8*)(Km + kro); kf0.h[1] = *(const ush8*)(Km + kro + 8);
    kf1.h[0] = *(const ush8*)(Km + kr1); kf1.h[1] = *(const ush8*)(Km + kr1 + 8);

    if (kb + 32 < kend){   // stream prefetch of next K/V tiles (global_prefetch_b8)
      __builtin_prefetch(Km + kro + 32*LATc, 0, 1);
      __builtin_prefetch(vrow0 + kb + 32 + bK0, 0, 1);
    }

    // qa shared by the S0/S1 pair -> reuse-A hint on the first
    v8f S0 = wmma_bf16<true,  false>(qa.v, kf0.v, vzero);
    v8f S1 = wmma_bf16<false, false>(qa.v, kf1.v, vzero);

    float mk0 = mkv ? mkv[b*nK + kb +      bN] : 1.0f;
    float mk1 = mkv ? mkv[b*nK + kb + 16 + bN] : 1.0f;

    #pragma unroll
    for (int r = 0; r < 8; ++r){
      float s0 = (mk0 != 0.0f) ? S0[r]*scale : -1.0e10f;
      float s1 = (mk1 != 0.0f) ? S1[r]*scale : -1.0e10f;
      float rm = fmaxf(s0, s1);
      #pragma unroll
      for (int d = 1; d < 16; d <<= 1) rm = fmaxf(rm, __shfl_xor(rm, d, 16));
      float mnew = fmaxf(mrow[r], rm);
      float p0 = __expf(s0 - mnew);
      float p1 = __expf(s1 - mnew);
      float rs = p0 + p1;
      #pragma unroll
      for (int d = 1; d < 16; d <<= 1) rs += __shfl_xor(rs, d, 16);
      float co = __expf(mrow[r] - mnew);
      lrow[r] = lrow[r]*co + rs;
      mrow[r] = mnew;
      acc0[r] *= co;
      acc1[r] *= co;
      int M = r + 8*(lane >> 4);
      sP[wave][M][bN]      = f2bf(p0);
      sP[wave][M][16 + bN] = f2bf(p1);
    }
    V16u pa;
    pa.h[0] = *(const ush8*)&sP[wave][lane & 15][aK0];
    pa.h[1] = *(const ush8*)&sP[wave][lane & 15][aK0 + 8];

    V16u vf0, vf1;
    vf0.h[0] = *(const ush8*)(vrow0 + kb + bK0);
    vf0.h[1] = *(const ush8*)(vrow0 + kb + bK0 + 8);
    vf1.h[0] = *(const ush8*)(vrow1 + kb + bK0);
    vf1.h[1] = *(const ush8*)(vrow1 + kb + bK0 + 8);

    // pa shared by the PV pair -> reuse-A hint on the first
    acc0 = wmma_bf16<true,  false>(pa.v, vf0.v, acc0);
    acc1 = wmma_bf16<false, false>(pa.v, vf1.v, acc1);
  }

  if (pacc){
    // split-K partials: unnormalized acc + per-row (m, l)
    #pragma unroll
    for (int r = 0; r < 8; ++r){
      int q = qt*128 + wave*16 + r + 8*(lane >> 4);
      size_t pb = (((size_t)(split*Bc + b)*NHc + h)*128 + q)*32;
      pacc[pb + bN]      = acc0[r];
      pacc[pb + 16 + bN] = acc1[r];
      if (bN == 0){
        size_t mb = (((size_t)(split*Bc + b)*NHc + h)*128 + q)*2;
        pml[mb]     = mrow[r];
        pml[mb + 1] = lrow[r];
      }
    }
  } else {
    #pragma unroll
    for (int r = 0; r < 8; ++r){
      int M = r + 8*(lane >> 4);
      int q = qt*128 + wave*16 + M;
      size_t ob = ((size_t)(b*nQ + q)) * LATc + h*DHc;
      float inv = 1.0f / lrow[r];
      float o0 = acc0[r]*inv + bf2f(Q[ob + bN]);
      float o1 = acc1[r]*inv + bf2f(Q[ob + 16 + bN]);
      O[ob + bN]      = f2bf(o0);
      O[ob + 16 + bN] = f2bf(o1);
    }
  }
}

// ---------------- split-K combine (log-sum-exp merge + Q residual) ----------------
__global__ __launch_bounds__(256) void attn_combine_kernel(
    const float* __restrict__ pacc, const float* __restrict__ pml,
    const unsigned short* __restrict__ Q, unsigned short* __restrict__ O,
    int nsplit)
{
  int blk = blockIdx.x;            // b*NHc + h
  int h = blk & 3, b = blk >> 2;
  for (int e = threadIdx.x; e < 128*32; e += 256){
    int q = e >> 5, d = e & 31;
    float M = -3.0e38f;
    for (int s = 0; s < nsplit; ++s){
      size_t mb = (((size_t)(s*Bc + b)*NHc + h)*128 + q)*2;
      M = fmaxf(M, pml[mb]);
    }
    float L = 0.0f, acc = 0.0f;
    for (int s = 0; s < nsplit; ++s){
      size_t mb = (((size_t)(s*Bc + b)*NHc + h)*128 + q)*2;
      float w = __expf(pml[mb] - M);
      L   += pml[mb + 1] * w;
      acc += pacc[(((size_t)(s*Bc + b)*NHc + h)*128 + q)*32 + d] * w;
    }
    size_t ob = ((size_t)(b*128 + q))*LATc + h*DHc + d;
    float o = acc / L + bf2f(Q[ob]);
    O[ob] = f2bf(o);
  }
}

// ---------------- host orchestration ----------------
extern "C" void kernel_launch(void* const* d_in, const int* in_sizes, int n_in,
                              void* d_out, int out_size, void* d_ws, size_t ws_size,
                              hipStream_t stream)
{
  (void)in_sizes; (void)n_in; (void)out_size; (void)ws_size;
  const float* time_x  = (const float*)d_in[0];
  const float* value_x = (const float*)d_in[1];
  const int*   mask_x  = (const int*)  d_in[2];
  const float* Wi      = (const float*)d_in[3];
  const float* bi      = (const float*)d_in[4];
  const float* If      = (const float*)d_in[5];
  const float* Wq      = (const float*)d_in[6];
  const float* bq      = (const float*)d_in[7];
  const float* Wk      = (const float*)d_in[8];
  const float* bk      = (const float*)d_in[9];
  const float* Wv      = (const float*)d_in[10];
  const float* bv      = (const float*)d_in[11];
  const float* Wo      = (const float*)d_in[12];
  const float* bo      = (const float*)d_in[13];

  float* Zf = (float*)d_out;                     // Z master accumulator (fp32)
  float* mk = Zf + (size_t)Bc * Lc * LATc;       // mk lives in d_out tail

  char* wbase = (char*)d_ws;
  size_t off = 0;
  auto alloc = [&](size_t bytes)->char* {
    char* p = wbase + off;
    off = (off + bytes + 255) & ~(size_t)255;
    return p;
  };
  const size_t bigN = (size_t)Bc * Lc * LATc;
  const size_t smlN = (size_t)Bc * 128 * LATc;
  unsigned short* wI  = (unsigned short*)alloc((size_t)3*128*128*2);      // linear (A-side)
  unsigned short* wQ  = (unsigned short*)alloc((size_t)3*2*128*128*2);    // frag order
  unsigned short* wK  = (unsigned short*)alloc((size_t)3*2*128*128*2);
  unsigned short* wV  = (unsigned short*)alloc((size_t)3*2*128*128*2);
  unsigned short* wO  = (unsigned short*)alloc((size_t)3*2*128*128*2);
  int*            idx = (int*)           alloc((size_t)Bc*Lc*4);
  unsigned short* Zb   = (unsigned short*)alloc(bigN*2);  // bf16 shadow of Z
  unsigned short* Abig = (unsigned short*)alloc(bigN*2);  // K0 then Q1
  unsigned short* Bbig = (unsigned short*)alloc(bigN*2);  // Vt0 then Opre1
  unsigned short* Qs   = (unsigned short*)alloc(smlN*2);  // Qh0
  unsigned short* Ks   = (unsigned short*)alloc(smlN*2);  // Kh1
  unsigned short* Vs   = (unsigned short*)alloc(smlN*2);  // Vt1
  unsigned short* Os   = (unsigned short*)alloc(smlN*2);  // Opre0
  unsigned short* Hb   = (unsigned short*)alloc(smlN*2);  // H
  float* pacc = (float*)alloc((size_t)KSPLIT*Bc*NHc*128*32*4);  // split partials
  float* pml  = (float*)alloc((size_t)KSPLIT*Bc*NHc*128*2*4);   // split (m,l)

  const int Rbig = Bc * Lc;      // 167936 rows (multiple of 128)
  const int Rsml = Bc * 128;     // 2048 rows
  const int kChunk = Lc / KSPLIT;        // 1312, multiple of 32

  zero_f32_kernel<<<(Bc*Lc + 255)/256, 256, 0, stream>>>(mk, (long)Bc*Lc);
  compact_kernel<<<Bc, 256, 0, stream>>>(mask_x, idx, mk);
  {
    long nI = (long)3*128*128, nW = (long)3*2*128*128;
    f32_to_bf16_kernel<<<(int)((nI+255)/256), 256, 0, stream>>>(If, wI, nI);
    swizzle_w_kernel<<<(int)((nW+255)/256), 256, 0, stream>>>(Wq, wQ, nW);
    swizzle_w_kernel<<<(int)((nW+255)/256), 256, 0, stream>>>(Wk, wK, nW);
    swizzle_w_kernel<<<(int)((nW+255)/256), 256, 0, stream>>>(Wv, wV, nW);
    swizzle_w_kernel<<<(int)((nW+255)/256), 256, 0, stream>>>(Wo, wO, nW);
  }
  embed_kernel<<<Bc*Lc, 128, 0, stream>>>(time_x, value_x, idx, mk, Wi, bi, Zf, Zb);

  const int IDmask = 0x7fffffff;

  for (int l = 0; l < 3; ++l){
    const unsigned short* wq0 = wQ + ((size_t)l*2 + 0)*128*128;
    const unsigned short* wq1 = wQ + ((size_t)l*2 + 1)*128*128;
    const unsigned short* wk0 = wK + ((size_t)l*2 + 0)*128*128;
    const unsigned short* wk1 = wK + ((size_t)l*2 + 1)*128*128;
    const unsigned short* wv0 = wV + ((size_t)l*2 + 0)*128*128;
    const unsigned short* wv1 = wV + ((size_t)l*2 + 1)*128*128;
    const unsigned short* wo0 = wO + ((size_t)l*2 + 0)*128*128;
    const unsigned short* wo1 = wO + ((size_t)l*2 + 1)*128*128;
    const float* bq0 = bq + ((size_t)l*2 + 0)*128;
    const float* bq1 = bq + ((size_t)l*2 + 1)*128;
    const float* bk0 = bk + ((size_t)l*2 + 0)*128;
    const float* bk1 = bk + ((size_t)l*2 + 1)*128;
    const float* bv0 = bv + ((size_t)l*2 + 0)*128;
    const float* bv1 = bv + ((size_t)l*2 + 1)*128;
    const float* bo0 = bo + ((size_t)l*2 + 0)*128;
    const float* bo1 = bo + ((size_t)l*2 + 1)*128;
    const unsigned short* Il = wI + (size_t)l*128*128;

    // ---- MAB0: induced points attend to masked data ----
    gemm128_kernel<<<Rsml/128, 256, 0, stream>>>(Il, 127, wq0, bq0, 0, Qs,
                                                 nullptr, nullptr, nullptr, Rsml, 0);
    gemm128_kernel<<<Rbig/128, 256, 0, stream>>>(Zb, IDmask, wk0, bk0, 0, Abig,
                                                 nullptr, nullptr, nullptr, Rbig, 0);
    gemm128_kernel<<<Rbig/128, 256, 0, stream>>>(Zb, IDmask, wv0, bv0, 0, Bbig,
                                                 nullptr, nullptr, nullptr, Rbig, Lc);
    attn_kernel<<<KSPLIT*Bc*NHc, 256, 0, stream>>>(Qs, Abig, Bbig, mk, nullptr,
                                                   pacc, pml, 128, Lc, 1, kChunk);
    attn_combine_kernel<<<Bc*NHc, 256, 0, stream>>>(pacc, pml, Qs, Os, KSPLIT);
    gemm128_kernel<<<Rsml/128, 256, 0, stream>>>(Os, IDmask, wo0, bo0, 1, Hb,
                                                 Os, nullptr, nullptr, Rsml, 0);

    // ---- MAB1: data attends to induced points ----
    gemm128_kernel<<<Rbig/128, 256, 0, stream>>>(Zb, IDmask, wq1, bq1, 0, Abig,
                                                 nullptr, nullptr, nullptr, Rbig, 0);
    gemm128_kernel<<<Rsml/128, 256, 0, stream>>>(Hb, IDmask, wk1, bk1, 0, Ks,
                                                 nullptr, nullptr, nullptr, Rsml, 0);
    gemm128_kernel<<<Rsml/128, 256, 0, stream>>>(Hb, IDmask, wv1, bv1, 0, Vs,
                                                 nullptr, nullptr, nullptr, Rsml, 128);
    attn_kernel<<<Bc*(Lc/128)*NHc, 256, 0, stream>>>(Abig, Ks, Vs, nullptr, Bbig,
                                                     nullptr, nullptr, Lc, 128,
                                                     Lc/128, 128);
    gemm128_kernel<<<Rbig/128, 256, 0, stream>>>(Bbig, IDmask, wo1, bo1, 2, Zb,
                                                 Bbig, mk, Zf, Rbig, 0);
  }
}
